// SubsetsDKNN_35450660061327
// MI455X (gfx1250) — matrix operand order repaired and checked by reading, not compiled, not run on hardware
//
#include <hip/hip_runtime.h>

// SubsetsDKNN fused kernel for MI455X (gfx1250, wave32).
//
//   scores = -cdist(query, neighbors); w = scores + gumbel
//   10x: w += log(max(1-onehot,tiny)); onehot = softmax(w); khot += onehot
//
// Strategy (reasoned from MI455X specs):
//  * GEMM in f16 WMMA (v_wmma_f32_16x16x32_f16), norms exact in f32.
//  * Exp-domain reformulation: y=exp(w); per iter: p=y/S; khot+=p; y*=max(1-p,tiny).
//    Exactly equivalent to the softmax recursion at TAU=1; removes all
//    per-iteration transcendentals (one exp + one sqrt per element, ever).
//  * Fully fused: y resident in 256KB LDS (8 rows x 8192 f32) per 1024-thread
//    workgroup; khot in 64 VGPRs/lane; w never hits HBM.
//    HBM traffic ~= 268MB => ~11.5us floor at 23.3 TB/s.
//  * Neighbors (4MB f16) stay L2-resident (192MB L2) across the 512 WGs.
//  * Round-1 disasm fixes: preload all 8 B fragments (one 16x b128 clause ->
//    MLP instead of s_wait_loadcnt 0 per WMMA); hardware v_sqrt/v_rcp instead
//    of libm expansions; all 32 lanes active in the epilogue via shfl_xor of
//    C rows 4..7; float4 LDS traffic in the iteration loop (4x fewer ds ops).

#define QN   4096
#define NN   8192
#define DD   256
#define KSEL 10
#define NTILES (NN / 16)  // 512
// TAU == 1.0: softmax(w/TAU) == softmax(w), hardcoded below.

typedef _Float16 v16h __attribute__((ext_vector_type(16)));
typedef _Float16 h8   __attribute__((ext_vector_type(8)));
typedef float    v8f  __attribute__((ext_vector_type(8)));

__device__ __forceinline__ float wave_sum32(float v) {
  // wave32 butterfly reduction
  v += __shfl_xor(v, 16, 32);
  v += __shfl_xor(v, 8, 32);
  v += __shfl_xor(v, 4, 32);
  v += __shfl_xor(v, 2, 32);
  v += __shfl_xor(v, 1, 32);
  return v;
}

// f32 rows -> f16 rows + f32 sum-of-squares per row. One wave32 per row.
__global__ void __launch_bounds__(256) subsets_prep(const float* __restrict__ src,
                                                    _Float16* __restrict__ dsth,
                                                    float* __restrict__ dstsq) {
  const int row  = blockIdx.x * 8 + (threadIdx.x >> 5);
  const int lane = threadIdx.x & 31;
  const float* s = src + (size_t)row * DD + lane * 8;
  float ss = 0.0f;
  h8 h;
#pragma unroll
  for (int j = 0; j < 8; ++j) {
    float v = s[j];
    ss += v * v;
    h[j] = (_Float16)v;
  }
  *(h8*)(dsth + (size_t)row * DD + lane * 8) = h;  // 16B store
  ss = wave_sum32(ss);
  if (lane == 0) dstsq[row] = ss;
}

// Reduce acc[8] (per-thread partial per row) across the 1024-thread block
// into rowS[8]. red = [32 waves][8 rows] scratch.
__device__ __forceinline__ void block_row_sums(const float acc[8], float* red,
                                               float* rowS, int tid, int lane,
                                               int wave) {
#pragma unroll
  for (int r = 0; r < 8; ++r) {
    float v = wave_sum32(acc[r]);
    if (lane == 0) red[wave * 8 + r] = v;
  }
  __syncthreads();
  if (tid < 8) {
    float s = 0.0f;
    for (int w = 0; w < 32; ++w) s += red[w * 8 + tid];
    rowS[tid] = s;
  }
  __syncthreads();
}

// One workgroup = 8 query rows x all 8192 neighbors, fully fused.
__global__ void __launch_bounds__(1024) subsets_fused(
    const _Float16* __restrict__ qh, const _Float16* __restrict__ nh,
    const float* __restrict__ q2, const float* __restrict__ n2,
    const float* __restrict__ gum, float* __restrict__ out) {
  extern __shared__ float smem[];
  float* yb   = smem;           // [8][8192] y = exp(w), f32
  float* red  = smem + 8 * NN;  // [32][8] reduction scratch
  float* rowS = red + 32 * 8;   // [8] per-row sums

  const int tid  = threadIdx.x;
  const int lane = tid & 31;
  const int wave = tid >> 5;
  const int m0   = blockIdx.x * 8;
  const int hi   = lane >> 4;  // 0: lanes 0-15, 1: lanes 16-31

  // Exact f32 query norms (uniform -> scalar loads). Pre-select the 4 rows
  // this lane-half handles in the epilogue (constant indices only).
  float q2r[8];
#pragma unroll
  for (int r = 0; r < 8; ++r) q2r[r] = q2[m0 + r];
  float q2v[4];
#pragma unroll
  for (int j = 0; j < 4; ++j) q2v[j] = hi ? q2r[4 + j] : q2r[j];

  // ---- Preload A fragments: 8 query rows padded to the 16-row WMMA tile ----
  // ISA 16-bit A 16x32 layout: lane l holds row M=l&15; lanes 0-15 carry
  // K {0..7, 16..23}, lanes 16-31 carry K {8..15, 24..31} (2 f16 / VGPR).
  const int mi = lane & 15;
  const int kh = hi * 8;
  v16h afr[8];
#pragma unroll
  for (int ks = 0; ks < 8; ++ks) {
    v16h a;
    if (mi < 8) {
      const _Float16* ap = qh + (size_t)(m0 + mi) * DD + ks * 32 + kh;
      h8 lo = *(const h8*)ap;         // K kb..kb+7
      h8 hx = *(const h8*)(ap + 16);  // K kb+16..kb+23
      a = __builtin_shufflevector(lo, hx, 0, 1, 2, 3, 4, 5, 6, 7, 8, 9, 10, 11,
                                  12, 13, 14, 15);
    } else {
#pragma unroll
      for (int j = 0; j < 16; ++j) a[j] = (_Float16)0.0f;  // pad rows 8..15
    }
    afr[ks] = a;
  }

  float sacc[4];  // per-lane partial y-sums for the 4 rows this half handles
#pragma unroll
  for (int j = 0; j < 4; ++j) sacc[j] = 0.0f;

  // ---- GEMM + epilogue: 512 N-tiles of 16, striped across 32 waves ----
  // B 32x16 layout: lane l = column N=l&15; lanes 0-15 K {0..7, 16..23}, 16-31 K {8..15, 24..31}.
  const int bkh = hi * 8;
  const int rbase = hi * 4;  // rows this lane-half owns in the epilogue
  for (int t = wave; t < NTILES; t += 32) {
    const int n = t * 16 + (lane & 15);
    const _Float16* bp = nh + (size_t)n * DD + bkh;

    // Independent epilogue operands: issue early, consume after the WMMAs.
    const float n2v = n2[n];
    float g[4];
#pragma unroll
    for (int j = 0; j < 4; ++j)
      g[j] = gum[(size_t)(m0 + rbase + j) * NN + n];

    // All 8 B fragments up front: 16 b128 loads in flight per wave.
    v16h bf[8];
#pragma unroll
    for (int ks = 0; ks < 8; ++ks) {
      h8 lo = *(const h8*)(bp + ks * 32);
      h8 hx = *(const h8*)(bp + ks * 32 + 16);
      bf[ks] = __builtin_shufflevector(lo, hx, 0, 1, 2, 3, 4, 5, 6, 7, 8, 9,
                                       10, 11, 12, 13, 14, 15);
    }
    // Uniform-clamped prefetch of this wave's next tile (s_cselect, no exec).
    const int tp = (t + 32 < NTILES) ? (t + 32) : t;
    __builtin_prefetch(nh + ((size_t)tp * 16 + (lane & 15)) * DD + bkh, 0, 1);

    v8f c = {};
#pragma unroll
    for (int ks = 0; ks < 8; ++ks)
      c = __builtin_amdgcn_wmma_f32_16x16x32_f16(false, afr[ks], false, bf[ks],
                                                 (short)0, c, false, false);

    // C layout: element r, lane l -> M = r + 8*(l>=16), N = l&15; valid rows
    // 0..7 live in lanes 0..15. Swap rows 4..7 to the upper half so all 32
    // lanes run the epilogue (4 rows each).
    float cs[4];
#pragma unroll
    for (int j = 0; j < 4; ++j) cs[j] = __shfl_xor(c[4 + j], 16, 32);
#pragma unroll
    for (int j = 0; j < 4; ++j) {
      const float dot = hi ? cs[j] : c[j];
      const float d2 = fmaxf(q2v[j] + n2v - 2.0f * dot, 0.0f);
      const float wv = g[j] - __builtin_amdgcn_sqrtf(d2);
      const float yv = __expf(wv);  // the only transcendental per element
      yb[(rbase + j) * NN + n] = yv;
      sacc[j] += yv;
    }
  }

  // Scatter the half-specific partials into the 8-row accumulator layout.
  float acc8[8];
#pragma unroll
  for (int j = 0; j < 4; ++j) {
    acc8[j]     = hi ? 0.0f : sacc[j];
    acc8[4 + j] = hi ? sacc[j] : 0.0f;
  }

  // khot accumulators: 16 float4 (64 elements)/thread; float4 slot
  // e4 -> flat4 = e4*1024 + tid, row = e4>>1 (constant per e4).
  float4 khot[16];
#pragma unroll
  for (int e = 0; e < 16; ++e) khot[e] = make_float4(0.f, 0.f, 0.f, 0.f);

  block_row_sums(acc8, red, rowS, tid, lane, wave);  // S_0 = sum_n y
  float4* yb4 = (float4*)yb;

  // ---- 10 relaxed top-k iterations, pure mul/add over LDS-resident y ----
  for (int it = 0; it < KSEL; ++it) {
    float rinv[8];
#pragma unroll
    for (int r = 0; r < 8; ++r)
      rinv[r] = __builtin_amdgcn_rcpf(fmaxf(rowS[r], 1e-38f));
    float np[8];
#pragma unroll
    for (int r = 0; r < 8; ++r) np[r] = 0.0f;
#pragma unroll
    for (int e = 0; e < 16; ++e) {
      const int idx = e * 1024 + tid;  // thread-private LDS slots: no races
      float4 y4 = yb4[idx];            // ds_load_b128
      const float ri = rinv[e >> 1];
      const float px = y4.x * ri, py = y4.y * ri, pz = y4.z * ri,
                  pw = y4.w * ri;      // onehot = softmax(w)
      khot[e].x += px; khot[e].y += py; khot[e].z += pz; khot[e].w += pw;
      // w += log(max(1-p, tiny)) in exp domain:
      y4.x *= fmaxf(1.0f - px, 1.17549435e-38f);
      y4.y *= fmaxf(1.0f - py, 1.17549435e-38f);
      y4.z *= fmaxf(1.0f - pz, 1.17549435e-38f);
      y4.w *= fmaxf(1.0f - pw, 1.17549435e-38f);
      yb4[idx] = y4;                   // ds_store_b128
      np[e >> 1] += (y4.x + y4.y) + (y4.z + y4.w);  // next iteration's sum
    }
    block_row_sums(np, red, rowS, tid, lane, wave);
  }

  // ---- write khot (float4, fully coalesced b128 stores) ----
  float4* out4 = (float4*)(out + (size_t)m0 * NN);
#pragma unroll
  for (int e = 0; e < 16; ++e) out4[e * 1024 + tid] = khot[e];
}

extern "C" void kernel_launch(void* const* d_in, const int* in_sizes, int n_in,
                              void* d_out, int out_size, void* d_ws,
                              size_t ws_size, hipStream_t stream) {
  (void)in_sizes; (void)n_in; (void)out_size; (void)ws_size;
  const float* query = (const float*)d_in[0];  // [4096,256] f32
  const float* neigh = (const float*)d_in[1];  // [1,8192,256] f32
  const float* gum   = (const float*)d_in[2];  // [4096,8192] f32
  float* out = (float*)d_out;                  // [4096,8192] f32

  // Workspace layout (~6.05 MB): qh f16 | nh f16 | q2 f32 | n2 f32
  char* ws = (char*)d_ws;
  _Float16* qh = (_Float16*)ws;
  _Float16* nh = (_Float16*)(ws + (size_t)QN * DD * sizeof(_Float16));
  float* q2 = (float*)(ws + (size_t)(QN + NN) * DD * sizeof(_Float16));
  float* n2 = q2 + QN;

  subsets_prep<<<QN / 8, 256, 0, stream>>>(query, qh, q2);
  subsets_prep<<<NN / 8, 256, 0, stream>>>(neigh, nh, n2);

  // 8 rows x 8192 f32 y-buffer + reduction scratch: ~257 KB dynamic LDS
  // (within CDNA5's 320 KB per-workgroup LDS budget).
  const size_t lds_bytes = (size_t)(8 * NN + 32 * 8 + 8) * sizeof(float);
  subsets_fused<<<QN / 8, 1024, lds_bytes, stream>>>(qh, nh, q2, n2, gum, out);
}
